// vMFLunaBlock_45603962749211
// MI455X (gfx1250) — compile-verified, matrix operand order checked by reading, not run on hardware
//
#include <hip/hip_runtime.h>
#include <math.h>

typedef unsigned short u16;
typedef __attribute__((ext_vector_type(8)))  u16    u16x8;
typedef __attribute__((ext_vector_type(16))) u16    u16x16;
typedef __attribute__((ext_vector_type(16))) __bf16 bf16x16;
typedef __attribute__((ext_vector_type(8)))  float  v8f;

// ---------------- scalar conversion helpers ----------------
__device__ __forceinline__ u16 f2bf(float f) {
    unsigned u = __builtin_bit_cast(unsigned, f);
    unsigned r = u + 0x7FFFu + ((u >> 16) & 1u);   // round-to-nearest-even
    return (u16)(r >> 16);
}
__device__ __forceinline__ float bf2f(u16 h) {
    unsigned u = ((unsigned)h) << 16;
    return __builtin_bit_cast(float, u);
}
__device__ __forceinline__ float gelu_exact(float x) {
    return 0.5f * x * (1.0f + erff(x * 0.70710678118654752f));
}

// ---------------- WMMA fragment helpers (wave32) ----------------
// 16-bit A/B fragment layout (CDNA5 ISA 7.12.2): lane<16 -> row = row0+lane,
// K = [k0, k0+8) in v0..3 and [k0+16, k0+24) in v4..7; lane>=16 -> K base +8.
__device__ __forceinline__ bf16x16 load_frag(const u16* __restrict__ m,
                                             int row0, int ld, int k0) {
    int lane = threadIdx.x & 31;
    const u16* p = m + (long)(row0 + (lane & 15)) * ld + k0 + ((lane >> 4) << 3);
    u16x8 lo = *(const u16x8*)p;
    u16x8 hi = *(const u16x8*)(p + 16);
    u16x16 v = __builtin_shufflevector(lo, hi, 0,1,2,3,4,5,6,7,8,9,10,11,12,13,14,15);
    return __builtin_bit_cast(bf16x16, v);
}

__device__ __forceinline__ v8f wmma_bf16(bf16x16 a, bf16x16 b, v8f c) {
    return __builtin_amdgcn_wmma_f32_16x16x32_bf16(
        /*neg_a=*/false, a, /*neg_b=*/false, b,
        /*c_mod=*/(short)0, c, /*reuse_a=*/false, /*reuse_b=*/false);
}

// ---------------- fp32 -> bf16 convert ----------------
__global__ __launch_bounds__(256) void cvt_bf16(const float* __restrict__ in,
                                                u16* __restrict__ out, long n) {
    long i = (long)blockIdx.x * 256 + threadIdx.x;
    if (i < n) out[i] = f2bf(in[i]);
}

// ---------------- batched GEMM: C = alpha * A(MxK) * B(NxK)^T ----------------
// A,B bf16 row-major (K contiguous).  batch z: zo=z/Hs, zi=z%Hs,
// operand offset = zo*Out + zi*In (elements).
// Block tile BM x BN, 256 threads (8 waves, 4 row-strips x 2 col-strips).
// Each wave computes (BM/4) x (BN/2) = RT x CT tiles of 16x16, K-step 32.
template<int BM, int BN, bool WF32, bool WB16, bool GELU_>
__global__ __launch_bounds__(256) void gemm_bt(
    const u16* __restrict__ A, const u16* __restrict__ B,
    float* __restrict__ Cf, u16* __restrict__ Cb,
    int M, int N, int K, int lda, int ldb, int ldc,
    long long aOut, long long aIn, long long bOut, long long bIn,
    long long cOut, long long cIn, int Hs, float alpha)
{
    constexpr int RT = BM / 64;          // 16-row tiles per wave
    constexpr int CT = BN / 32;          // 16-col tiles per wave
    constexpr int AP = BM / 64;          // cooperative-load passes for A tile
    constexpr int BP = BN / 64;          // cooperative-load passes for B tile

    __shared__ u16 As[BM * 40];
    __shared__ u16 Bs[BN * 40];

    int z  = blockIdx.z;
    int zo = z / Hs, zi = z % Hs;
    A += zo * aOut + zi * aIn;
    B += zo * bOut + zi * bIn;
    long long coff = zo * cOut + zi * cIn;

    int row0 = blockIdx.y * BM;
    int col0 = blockIdx.x * BN;
    int tid  = threadIdx.x;
    int wave = tid >> 5;
    int wm = wave & 3;          // row strip:  wm * (BM/4)
    int wn = wave >> 2;         // col strip:  wn * (BN/2)
    int lr = tid >> 2;          // 0..63 tile row for cooperative loads
    int lc = (tid & 3) << 3;    // 0,8,16,24 tile col

    v8f acc[RT][CT] = {};

    for (int k0 = 0; k0 < K; k0 += 32) {
        u16x8 areg[AP], breg[BP];
#pragma unroll
        for (int i = 0; i < AP; ++i)
            areg[i] = *(const u16x8*)(A + (long)(row0 + lr + i * 64) * lda + k0 + lc);
#pragma unroll
        for (int i = 0; i < BP; ++i)
            breg[i] = *(const u16x8*)(B + (long)(col0 + lr + i * 64) * ldb + k0 + lc);
        if (k0 + 32 < K) {
            __builtin_prefetch(A + (long)(row0 + lr) * lda + k0 + 32 + lc, 0, 1);
            __builtin_prefetch(B + (long)(col0 + lr) * ldb + k0 + 32 + lc, 0, 1);
        }
        __syncthreads();
#pragma unroll
        for (int i = 0; i < AP; ++i) *(u16x8*)&As[(lr + i * 64) * 40 + lc] = areg[i];
#pragma unroll
        for (int i = 0; i < BP; ++i) *(u16x8*)&Bs[(lr + i * 64) * 40 + lc] = breg[i];
        __syncthreads();

        bf16x16 afr[RT], bfr[CT];
#pragma unroll
        for (int r = 0; r < RT; ++r)
            afr[r] = load_frag(As, wm * (BM / 4) + r * 16, 40, 0);
#pragma unroll
        for (int c = 0; c < CT; ++c)
            bfr[c] = load_frag(Bs, wn * (BN / 2) + c * 16, 40, 0);
#pragma unroll
        for (int r = 0; r < RT; ++r)
#pragma unroll
            for (int c = 0; c < CT; ++c)
                acc[r][c] = wmma_bf16(afr[r], bfr[c], acc[r][c]);
    }

    int lane = tid & 31;
#pragma unroll
    for (int r = 0; r < RT; ++r) {
        int rb = row0 + wm * (BM / 4) + r * 16 + ((lane >> 4) << 3);
#pragma unroll
        for (int c = 0; c < CT; ++c) {
            int cc = col0 + wn * (BN / 2) + c * 16 + (lane & 15);
#pragma unroll
            for (int e = 0; e < 8; ++e) {
                float v = acc[r][c][e] * alpha;
                if (GELU_) v = gelu_exact(v);
                long long idx = coff + (long long)(rb + e) * ldc + cc;
                if (WF32) Cf[idx] = v;
                if (WB16) Cb[idx] = f2bf(v);
            }
        }
    }
}

// ---------------- grouped conv pool (kernel 4, stride 4) + bf16 ----------------
// lin: [B,S,H*Dh] bf16.  outn: [B,H,S4,Dh]  (K path)   outt: [B,H,Dh,S4] (V^T path)
__global__ __launch_bounds__(256) void conv_pool(
    const u16* __restrict__ lin, const float* __restrict__ w,
    u16* __restrict__ outn, u16* __restrict__ outt,
    int B_, int H_, int S_, int Dh)
{
    int S4 = S_ >> 2;
    long total = (long)B_ * H_ * S4 * Dh;
    long idx = (long)blockIdx.x * 256 + threadIdx.x;
    if (idx >= total) return;
    int d = idx % Dh;        long r = idx / Dh;
    int t = r % S4;          r /= S4;
    int h = r % H_;          int b = (int)(r / H_);
    int HD = H_ * Dh;
    const u16* p = lin + ((long)b * S_ + 4L * t) * HD + h * Dh + d;
    float acc = 0.f;
#pragma unroll
    for (int j = 0; j < 4; ++j) acc += w[h * 4 + j] * bf2f(p[(long)j * HD]);
    u16 o = f2bf(acc);
    long bh = (long)b * H_ + h;
    if (outn) outn[(bh * S4 + t) * Dh + d] = o;
    if (outt) outt[(bh * Dh + d) * (long)S4 + t] = o;
}

// ---------------- row softmax f32 -> bf16 probs ----------------
__global__ __launch_bounds__(256) void softmax_rows(const float* __restrict__ logits,
                                                    u16* __restrict__ probs, int cols) {
    long row = blockIdx.x;
    const float* src = logits + row * (long)cols;
    u16* dst = probs + row * (long)cols;
    __shared__ float red[256];
    int tid = threadIdx.x;

    float mx = -3.0e38f;
    for (int c = tid; c < cols; c += 256) mx = fmaxf(mx, src[c]);
    red[tid] = mx; __syncthreads();
    for (int s = 128; s > 0; s >>= 1) {
        if (tid < s) red[tid] = fmaxf(red[tid], red[tid + s]);
        __syncthreads();
    }
    mx = red[0]; __syncthreads();

    float sum = 0.f;
    for (int c = tid; c < cols; c += 256) sum += __expf(src[c] - mx);
    red[tid] = sum; __syncthreads();
    for (int s = 128; s > 0; s >>= 1) {
        if (tid < s) red[tid] += red[tid + s];
        __syncthreads();
    }
    float inv = 1.0f / red[0];
    for (int c = tid; c < cols; c += 256) dst[c] = f2bf(__expf(src[c] - mx) * inv);
}

// ---------------- split-heads transpose: [B,M,H*Dh] -> [B,H,Dh,M] ----------------
__global__ __launch_bounds__(256) void head_transpose(
    const u16* __restrict__ in, u16* __restrict__ out,
    int B_, int Mr, int H_, int Dh)
{
    long total = (long)B_ * H_ * Dh * Mr;
    long idx = (long)blockIdx.x * 256 + threadIdx.x;
    if (idx >= total) return;
    int m = idx % Mr;   long r = idx / Mr;
    int d = r % Dh;     r /= Dh;
    int h = r % H_;     int b = (int)(r / H_);
    out[idx] = in[((long)b * Mr + m) * (H_ * Dh) + h * Dh + d];
}

// ---------------- residual + mask + LayerNorm (D = 768) ----------------
__global__ __launch_bounds__(256) void ln_residual(
    const float* __restrict__ base, const float* __restrict__ add,
    const int* __restrict__ mask,
    float* __restrict__ outf, u16* __restrict__ outb, int D_)
{
    long row = blockIdx.x;
    float ms = mask ? (float)mask[row] : 1.0f;
    const float* bp = base + row * (long)D_;
    const float* ap = add  + row * (long)D_;
    int tid = threadIdx.x;
    float vals[3];
    float s = 0.f, ss = 0.f;
#pragma unroll
    for (int i = 0; i < 3; ++i) {
        int c = tid + i * 256;
        float v = bp[c] + ap[c] * ms;
        vals[i] = v; s += v; ss += v * v;
    }
    __shared__ float r1[256], r2[256];
    r1[tid] = s; r2[tid] = ss; __syncthreads();
    for (int st = 128; st > 0; st >>= 1) {
        if (tid < st) { r1[tid] += r1[tid + st]; r2[tid] += r2[tid + st]; }
        __syncthreads();
    }
    float mean = r1[0] / D_;
    float var  = r2[0] / D_ - mean * mean;
    float rstd = rsqrtf(var + 1e-6f);
    float* op = outf + row * (long)D_;
    u16*   ob = outb ? outb + row * (long)D_ : nullptr;
#pragma unroll
    for (int i = 0; i < 3; ++i) {
        int c = tid + i * 256;
        float o = (vals[i] - mean) * rstd;
        op[c] = o;
        if (ob) ob[c] = f2bf(o);
    }
}

// ---------------- fused unpack attention ----------------
// per block: one (b,h), 32 query rows, all 256 keys.  q: [B,S,768] bf16 (pre-scaled),
// k: [B,256,768] bf16, vT: [B,H,64,256] bf16, ctx out: [B,S,768] bf16.
__global__ __launch_bounds__(256) void attn_unpack(
    const u16* __restrict__ q, const u16* __restrict__ k,
    const u16* __restrict__ vT, u16* __restrict__ ctx,
    int S_, int Mk, int H_, int Dh)
{
    __shared__ u16   Ks[256 * 72];     // K tile  [256 x 64]
    __shared__ u16   Vt[64 * 264];     // V^T     [64 x 256]
    __shared__ float Sc[32 * 264];     // scores  [32 x 256] f32
    __shared__ u16   Pb[32 * 264];     // probs   [32 x 256] bf16
    __shared__ float redm[32 * 8];
    __shared__ float reds[32 * 8];

    int bh = blockIdx.y;
    int b = bh / H_, h = bh % H_;
    int qrow0 = blockIdx.x * 32;
    int tid = threadIdx.x;
    int HD = H_ * Dh;

    // stage K (256 rows x 64) : one row per thread
    {
        const u16* src = k + ((long)b * Mk + tid) * HD + h * Dh;
        u16* dst = &Ks[tid * 72];
#pragma unroll
        for (int j = 0; j < 8; ++j) ((u16x8*)dst)[j] = ((const u16x8*)src)[j];
    }
    // stage V^T (64 rows x 256) : 4 threads per row
    {
        int r = tid >> 2, c = (tid & 3) * 64;
        const u16* src = vT + ((long)bh * Dh + r) * Mk + c;
        u16* dst = &Vt[r * 264 + c];
#pragma unroll
        for (int j = 0; j < 8; ++j) ((u16x8*)dst)[j] = ((const u16x8*)src)[j];
    }
    __syncthreads();

    int wave = tid >> 5;
    int lane = tid & 31;
    int wm = wave & 1;       // 0..1 -> 16-row strip of the 32 q rows
    int wn = wave >> 1;      // 0..3 -> 64-col strip of the 256 keys

    // S = Q * K^T  (Q fragments straight from global)
    const u16* qbase = q + ((long)b * S_ + qrow0) * HD + h * Dh;
    v8f acc[4] = {};
    for (int k0 = 0; k0 < Dh; k0 += 32) {
        bf16x16 af = load_frag(qbase, wm * 16, HD, k0);
#pragma unroll
        for (int j = 0; j < 4; ++j) {
            bf16x16 bf = load_frag(Ks, wn * 64 + j * 16, 72, k0);
            acc[j] = wmma_bf16(af, bf, acc[j]);
        }
    }
    {
        int rb = wm * 16 + ((lane >> 4) << 3);
#pragma unroll
        for (int j = 0; j < 4; ++j) {
            int c = wn * 64 + j * 16 + (lane & 15);
#pragma unroll
            for (int r = 0; r < 8; ++r) Sc[(rb + r) * 264 + c] = acc[j][r];
        }
    }
    __syncthreads();

    // softmax over 256 keys: 8 threads per query row
    {
        int r = tid >> 3, seg = tid & 7, c0 = seg * 32;
        float mx = -3.0e38f;
        for (int c = 0; c < 32; ++c) mx = fmaxf(mx, Sc[r * 264 + c0 + c]);
        redm[r * 8 + seg] = mx; __syncthreads();
        if (seg == 0) {
            float m = redm[r * 8];
#pragma unroll
            for (int j = 1; j < 8; ++j) m = fmaxf(m, redm[r * 8 + j]);
            redm[r * 8] = m;
        }
        __syncthreads();
        mx = redm[r * 8];
        float sum = 0.f;
        for (int c = 0; c < 32; ++c) sum += __expf(Sc[r * 264 + c0 + c] - mx);
        reds[r * 8 + seg] = sum; __syncthreads();
        if (seg == 0) {
            float t = 0.f;
#pragma unroll
            for (int j = 0; j < 8; ++j) t += reds[r * 8 + j];
            reds[r * 8] = t;
        }
        __syncthreads();
        float inv = 1.0f / reds[r * 8];
        for (int c = 0; c < 32; ++c)
            Pb[r * 264 + c0 + c] = f2bf(__expf(Sc[r * 264 + c0 + c] - mx) * inv);
    }
    __syncthreads();

    // O = P * V   (V^T is [N=64, K=256] row-major -> same A*B^T path)
    v8f o = {};
    for (int k0 = 0; k0 < Mk; k0 += 32) {
        bf16x16 af = load_frag(Pb, wm * 16, 264, k0);
        bf16x16 bf = load_frag(Vt, wn * 16, 264, k0);
        o = wmma_bf16(af, bf, o);
    }
    u16* cb = ctx + ((long)b * S_ + qrow0) * HD + h * Dh;
    int rb = wm * 16 + ((lane >> 4) << 3);
    int c  = wn * 16 + (lane & 15);
#pragma unroll
    for (int r = 0; r < 8; ++r) cb[(long)(rb + r) * HD + c] = f2bf(o[r]);
}

// =====================================================================
extern "C" void kernel_launch(void* const* d_in, const int* in_sizes, int n_in,
                              void* d_out, int out_size, void* d_ws, size_t ws_size,
                              hipStream_t stream) {
    (void)in_sizes; (void)n_in; (void)out_size; (void)ws_size;
    const int B = 4, S = 4096, Mm = 256, D = 768, H = 12, Dh = 64, S4 = 1024, MLP = 3072;
    const int HD = H * Dh;                // 768

    const float* x    = (const float*)d_in[0];
    const float* mem  = (const float*)d_in[1];
    const int*   mask = (const int*)  d_in[2];
    const float* Wq = (const float*)d_in[3], *Wk = (const float*)d_in[4];
    const float* Wv = (const float*)d_in[5], *Wo = (const float*)d_in[6];
    const float* cw = (const float*)d_in[7];
    const float* Uq = (const float*)d_in[8], *Uk = (const float*)d_in[9];
    const float* Uv = (const float*)d_in[10], *Uo = (const float*)d_in[11];
    const float* W1 = (const float*)d_in[12], *W2 = (const float*)d_in[13];

    float* q_out = (float*)d_out;                       // [B,S,D]
    float* m_out = q_out + (long)B * S * D;             // [B,Mm,D]

    // ---- workspace bump allocator
    size_t off = 0;
    auto alloc = [&](size_t bytes) -> void* {
        off = (off + 255) & ~(size_t)255;
        void* p = (char*)d_ws + off;
        off += bytes;
        return p;
    };
    const long nX = (long)B * S * D, nM = (long)B * Mm * D, nW = (long)D * D, nW1 = (long)MLP * D;
    u16* xb   = (u16*)alloc(nX * 2);
    u16* mb   = (u16*)alloc(nM * 2);
    u16* wqb  = (u16*)alloc(nW * 2);  u16* wkb = (u16*)alloc(nW * 2);
    u16* wvb  = (u16*)alloc(nW * 2);  u16* wob = (u16*)alloc(nW * 2);
    u16* uqw  = (u16*)alloc(nW * 2);  u16* ukw = (u16*)alloc(nW * 2);
    u16* uvw  = (u16*)alloc(nW * 2);  u16* uow = (u16*)alloc(nW * 2);
    u16* w1b  = (u16*)alloc(nW1 * 2); u16* w2b = (u16*)alloc(nW1 * 2);
    u16* qlb  = (u16*)alloc(nM * 2);                       // memory @ Wq^T
    u16* klb  = (u16*)alloc(nX * 2);
    u16* vlb  = (u16*)alloc(nX * 2);
    u16* kpb  = (u16*)alloc((long)B * H * S4 * Dh * 2);    // [B,H,S4,Dh]
    u16* vptb = (u16*)alloc((long)B * H * Dh * S4 * 2);    // [B,H,Dh,S4]
    float* logits = (float*)alloc((long)B * H * Mm * S4 * 4);
    u16* probs = (u16*)alloc((long)B * H * Mm * S4 * 2);
    u16* ctxb  = (u16*)alloc(nM * 2);                      // merged heads [B,Mm,768]
    float* packedf = (float*)alloc(nM * 4);
    u16*   packedb = (u16*)alloc(nM * 2);
    u16* uqlb = (u16*)alloc(nX * 2);
    u16* uklb = (u16*)alloc(nM * 2);
    u16* uvlb = (u16*)alloc(nM * 2);
    u16* uvtb = (u16*)alloc((long)B * H * Dh * Mm * 2);    // [B,H,Dh,Mm]
    u16* uctx = (u16*)alloc(nX * 2);
    float* unpackedf = (float*)alloc(nX * 4);
    float* qo1f = (float*)alloc(nX * 4);
    u16*   qo1b = (u16*)alloc(nX * 2);
    u16*   hb   = (u16*)alloc((long)B * S * MLP * 2);
    float* yf   = (float*)alloc(nX * 4);

    auto cvt = [&](const float* src, u16* dst, long n) {
        cvt_bf16<<<dim3((unsigned)((n + 255) / 256)), 256, 0, stream>>>(src, dst, n);
    };
    cvt(x, xb, nX);   cvt(mem, mb, nM);
    cvt(Wq, wqb, nW); cvt(Wk, wkb, nW); cvt(Wv, wvb, nW); cvt(Wo, wob, nW);
    cvt(Uq, uqw, nW); cvt(Uk, ukw, nW); cvt(Uv, uvw, nW); cvt(Uo, uow, nW);
    cvt(W1, w1b, nW1); cvt(W2, w2b, nW1);

    // 128x128-tile launchers (all large GEMMs)
    auto g_f32 = [&](const u16* A, const u16* B_, float* Cf, int M, int N, int K,
                     int lda, int ldb, int ldc,
                     long long aO, long long aI, long long bO, long long bI,
                     long long cO, long long cI, int Hs, int batch, float alpha) {
        gemm_bt<128, 128, true, false, false><<<dim3(N / 128, M / 128, batch), 256, 0, stream>>>(
            A, B_, Cf, nullptr, M, N, K, lda, ldb, ldc, aO, aI, bO, bI, cO, cI, Hs, alpha);
    };
    auto g_b16 = [&](const u16* A, const u16* B_, u16* Cb, int M, int N, int K,
                     int lda, int ldb, int ldc,
                     long long aO, long long aI, long long bO, long long bI,
                     long long cO, long long cI, int Hs, int batch, float alpha) {
        gemm_bt<128, 128, false, true, false><<<dim3(N / 128, M / 128, batch), 256, 0, stream>>>(
            A, B_, nullptr, Cb, M, N, K, lda, ldb, ldc, aO, aI, bO, bI, cO, cI, Hs, alpha);
    };

    // ---- pack attention -------------------------------------------------
    g_b16(mb, wqb, qlb, B * Mm, D, D, D, D, D, 0,0,0,0,0,0, 1, 1, 1.0f);   // q
    g_b16(xb, wkb, klb, B * S, D, D, D, D, D, 0,0,0,0,0,0, 1, 1, 1.0f);    // k
    g_b16(xb, wvb, vlb, B * S, D, D, D, D, D, 0,0,0,0,0,0, 1, 1, 1.0f);    // v
    {
        long tot = (long)B * H * S4 * Dh;
        unsigned gb = (unsigned)((tot + 255) / 256);
        conv_pool<<<gb, 256, 0, stream>>>(klb, cw, kpb, nullptr, B, H, S, Dh);
        conv_pool<<<gb, 256, 0, stream>>>(vlb, cw, nullptr, vptb, B, H, S, Dh);
    }
    // logits[bh] = q[b,:,h] @ kp[bh]^T   (48 x 256 x 1024, K=64)
    g_f32(qlb, kpb, logits, Mm, S4, Dh, HD, Dh, S4,
          (long long)Mm * HD, Dh, (long long)H * S4 * Dh, (long long)S4 * Dh,
          (long long)H * Mm * S4, (long long)Mm * S4, H, B * H, 1.0f);
    softmax_rows<<<dim3(B * H * Mm), 256, 0, stream>>>(logits, probs, S4);
    // ctx[bh] = probs[bh] @ vpT[bh]^T -> merged-head [B,Mm,768] bf16  (N=64 -> 64x64 tile)
    gemm_bt<64, 64, false, true, false><<<dim3(Dh / 64, Mm / 64, B * H), 256, 0, stream>>>(
        probs, vptb, nullptr, ctxb, Mm, Dh, S4, S4, S4, HD,
        (long long)H * Mm * S4, (long long)Mm * S4,
        (long long)H * Dh * S4, (long long)Dh * S4,
        (long long)Mm * HD, Dh, H, 1.0f);
    // packed = ctx @ Wo^T  (f32 + bf16)
    gemm_bt<128, 128, true, true, false><<<dim3(D / 128, (B * Mm) / 128, 1), 256, 0, stream>>>(
        ctxb, wob, packedf, packedb, B * Mm, D, D, D, D, D, 0,0,0,0,0,0, 1, 1.0f);
    // m_out = LN(memory + packed)
    ln_residual<<<dim3(B * Mm), 256, 0, stream>>>(mem, packedf, nullptr, m_out, nullptr, D);

    // ---- unpack attention ----------------------------------------------
    g_b16(xb, uqw, uqlb, B * S, D, D, D, D, D, 0,0,0,0,0,0, 1, 1, 0.125f);   // uq * 1/sqrt(64)
    g_b16(packedb, ukw, uklb, B * Mm, D, D, D, D, D, 0,0,0,0,0,0, 1, 1, 1.0f);
    g_b16(packedb, uvw, uvlb, B * Mm, D, D, D, D, D, 0,0,0,0,0,0, 1, 1, 1.0f);
    {
        long tot = (long)B * H * Dh * Mm;
        head_transpose<<<dim3((unsigned)((tot + 255) / 256)), 256, 0, stream>>>(
            uvlb, uvtb, B, Mm, H, Dh);
    }
    attn_unpack<<<dim3(S / 32, B * H), 256, 0, stream>>>(uqlb, uklb, uvtb, uctx, S, Mm, H, Dh);
    // unpacked = uctx @ Uo^T (f32)
    g_f32(uctx, uow, unpackedf, B * S, D, D, D, D, D, 0,0,0,0,0,0, 1, 1, 1.0f);
    // q_out1 = LN(x + unpacked * mask)  -> f32 + bf16
    ln_residual<<<dim3(B * S), 256, 0, stream>>>(x, unpackedf, mask, qo1f, qo1b, D);

    // ---- FFN ------------------------------------------------------------
    gemm_bt<128, 128, false, true, true><<<dim3(MLP / 128, (B * S) / 128, 1), 256, 0, stream>>>(
        qo1b, w1b, nullptr, hb, B * S, MLP, D, D, D, MLP, 0,0,0,0,0,0, 1, 1.0f);
    g_f32(hb, w2b, yf, B * S, D, MLP, MLP, MLP, D, 0,0,0,0,0,0, 1, 1, 1.0f);
    // q_out = LN(q_out1 + y * mask)
    ln_residual<<<dim3(B * S), 256, 0, stream>>>(qo1f, yf, mask, q_out, nullptr, D);
}
